// RelativePositionTransformerEncoder_12326556139723
// MI455X (gfx1250) — matrix operand order repaired and checked by reading, not compiled, not run on hardware
//
#include <hip/hip_runtime.h>

typedef unsigned short u16;
typedef unsigned int u32;
typedef __bf16 bf16x16 __attribute__((ext_vector_type(16)));
typedef float f32x8 __attribute__((ext_vector_type(8)));
typedef u16 u16x8 __attribute__((ext_vector_type(8)));
typedef u32 u32x4 __attribute__((ext_vector_type(4)));
typedef int i32x4 __attribute__((ext_vector_type(4)));
typedef int i32x8 __attribute__((ext_vector_type(8)));

#define AS1 __attribute__((address_space(1)))
#define AS3 __attribute__((address_space(3)))

#define Bsz 8
#define Ssz 1024
#define Msz 1024
#define Hsz 8
#define Esz 128
#define Isz 2048
#define Lsz 6

#if __has_builtin(__builtin_amdgcn_tensor_load_to_lds)
#define HAS_TDM 1
#else
#define HAS_TDM 0
#endif

union FragU { u16x8 h[2]; bf16x16 v; };

__device__ inline u16 f2b(float f) {
  u32 x = __builtin_bit_cast(u32, f);
  u32 r = x + 0x7fffu + ((x >> 16) & 1u);
  return (u16)(r >> 16);
}

// Async 16B global -> LDS copy (ASYNCcnt) with synchronous fallback.
// Signature (from clang diagnostic): (AS1 v4i32*, AS3 v4i32*, imm, imm).
__device__ inline void async_ld_b128(const u16* g, u16* l) {
#if __has_builtin(__builtin_amdgcn_global_load_async_to_lds_b128)
  u16* gnc = const_cast<u16*>(g);
  __builtin_amdgcn_global_load_async_to_lds_b128((AS1 i32x4*)gnc,
                                                 (AS3 i32x4*)l, 0, 0);
#else
  *(u16x8*)l = *(const u16x8*)g;
#endif
}

__device__ inline void wait_async() {
#if __has_builtin(__builtin_amdgcn_s_wait_asynccnt)
  __builtin_amdgcn_s_wait_asynccnt(0);
#endif
}

// TDM: 1-D contiguous tile copy global -> LDS. bytes must be a multiple of 8
// and <= 512KB (tile_dim0 16-bit in 8B units). Issued per-wave (EXEC ignored).
__device__ inline void tdm_load_contig(const void* gsrc, void* ldst,
                                       unsigned bytes) {
#if HAS_TDM
  const unsigned elems = bytes >> 3;  // 8-byte units
  const u32 lds_off = (u32)(AS3 void*)ldst;
  const unsigned long long ga = (unsigned long long)gsrc;
  u32x4 g0;
  g0[0] = 1u;                                             // count=1
  g0[1] = lds_off;                                        // lds_addr
  g0[2] = (u32)ga;                                        // global_addr lo
  g0[3] = (u32)((ga >> 32) & 0x01FFFFFFu) | (2u << 30);   // addr hi | type=2
  i32x8 g1 = {};
  g1[0] = (3 << 16);                          // data_size = 8B
  g1[1] = (int)((elems & 0xFFFFu) << 16);     // tensor_dim0[15:0]
  g1[2] = (int)((elems >> 16) & 0xFFFFu);     // tensor_dim0[31:16]
  g1[3] = (int)(elems << 16);                 // tile_dim0 (1-D tile)
  i32x4 g2 = {};
  i32x4 g3 = {};
#if __has_include(<hip/amd_detail/amd_gfx1250_TDM.h>)
  i32x8 g4 = {};
  __builtin_amdgcn_tensor_load_to_lds(g0, g1, g2, g3, g4, 0);
#else
  __builtin_amdgcn_tensor_load_to_lds(g0, g1, g2, g3, 0);
#endif
#else
  (void)gsrc; (void)ldst; (void)bytes;
#endif
}

__device__ inline void wait_tensor() {
#if __has_builtin(__builtin_amdgcn_s_wait_tensorcnt)
  __builtin_amdgcn_s_wait_tensorcnt(0);
#endif
}

// A-operand fragment (16x32 bf16), ISA layout: lanes 0-15 hold row=lane,
// K {0..7,16..23}; lanes 16-31 hold row=lane-16, K {8..15,24..31}.
__device__ inline bf16x16 fragA(const u16* base, int stride, int kbase, int lane) {
  FragU f;
  const u16* p = base + (lane & 15) * stride + kbase + (lane >> 4) * 8;
  f.h[0] = *(const u16x8*)p;
  f.h[1] = *(const u16x8*)(p + 16);
  return f.v;
}

// B-operand fragment (32x16 bf16) read from a [N][K]-transposed LDS tile:
// lane holds col = lane&15, K {0..7,16..23} (lanes 0-15) / {8..15,24..31}.
__device__ inline bf16x16 fragB(const u16* base, int stride, int kbase, int lane) {
  FragU f;
  const u16* p = base + (lane & 15) * stride + kbase + (lane >> 4) * 8;
  f.h[0] = *(const u16x8*)p;
  f.h[1] = *(const u16x8*)(p + 16);
  return f.v;
}

__device__ inline f32x8 wmma_bf16(bf16x16 a, bf16x16 b, f32x8 c) {
  return __builtin_amdgcn_wmma_f32_16x16x32_bf16(false, a, false, b, (short)0, c,
                                                 false, false);
}

// ---------------------------------------------------------------------------
// Tiled bf16 GEMM: C[M,N] (fp32) = A[M,K] * Bt[N,K]^T, optional per-col bias.
// Both operands row-major with contiguous K; 128x128 block, K-step 32,
// 8 waves (2x4), each wave 64x32 of output. Async LDS staging when available.
// ---------------------------------------------------------------------------
__global__ __launch_bounds__(256) void gemm_bf16_kernel(
    const u16* __restrict__ A, const u16* __restrict__ Bt,
    float* __restrict__ C, const float* __restrict__ bias, int N, int K) {
  __shared__ u16 lsA[2][128][32];
  __shared__ u16 lsB[2][128][32];  // [n][k]
  const int tid = threadIdx.x, lane = tid & 31, wid = tid >> 5;
  const int wr = wid >> 2, wc = wid & 3;
  const size_t row0 = (size_t)blockIdx.y * 128;
  const int col0 = blockIdx.x * 128;
  f32x8 acc[4][2] = {};
  const int tRow = tid >> 1, tK = (tid & 1) << 4;

  auto stage = [&](int buf, int k0) {
    const u16* ga = A + (row0 + tRow) * (size_t)K + (k0 + tK);
    async_ld_b128(ga, &lsA[buf][tRow][tK]);
    async_ld_b128(ga + 8, &lsA[buf][tRow][tK + 8]);
    const u16* gb = Bt + (size_t)(col0 + tRow) * K + (k0 + tK);
    async_ld_b128(gb, &lsB[buf][tRow][tK]);
    async_ld_b128(gb + 8, &lsB[buf][tRow][tK + 8]);
    __builtin_prefetch(ga + 32, 0, 1);
    __builtin_prefetch(gb + 32, 0, 1);
  };

  stage(0, 0);
  wait_async();
  __syncthreads();
  const int nk = K >> 5;
  for (int kb = 0; kb < nk; ++kb) {
    const int buf = kb & 1;
    if (kb + 1 < nk) stage(buf ^ 1, (kb + 1) << 5);
#pragma unroll
    for (int i = 0; i < 4; ++i) {
      bf16x16 a = fragA(&lsA[buf][wr * 64 + i * 16][0], 32, 0, lane);
#pragma unroll
      for (int j = 0; j < 2; ++j) {
        bf16x16 b = fragB(&lsB[buf][wc * 32 + j * 16][0], 32, 0, lane);
        acc[i][j] = wmma_bf16(a, b, acc[i][j]);
      }
    }
    wait_async();
    __syncthreads();
  }
  const int rHi = (lane >> 4) * 8;
  const int cLane = lane & 15;
#pragma unroll
  for (int i = 0; i < 4; ++i)
#pragma unroll
    for (int j = 0; j < 2; ++j)
#pragma unroll
      for (int r = 0; r < 8; ++r) {
        const size_t row = row0 + wr * 64 + i * 16 + r + rHi;
        const int col = col0 + wc * 32 + j * 16 + cLane;
        float v = acc[i][j][r];
        if (bias) v += bias[col];
        C[row * N + col] = v;
      }
}

// ---------------------------------------------------------------------------
// Flash attention with factorized relative-position term.
// Qc/Kc: [B,H,S,256] bf16 (content ++ positional half), Vh: [B,H,S,128] bf16.
// One block = (64 q-rows, one b, one h); 4 waves, each wave 16 q-rows.
// K tiles staged via TDM (when present), Q via async-to-LDS.
// ---------------------------------------------------------------------------
__global__ __launch_bounds__(128) void attn_kernel(
    const u16* __restrict__ Qc, const u16* __restrict__ Kc,
    const u16* __restrict__ Vh, const unsigned char* __restrict__ mask,
    u16* __restrict__ O) {
  __shared__ u16 lsQ[64][256];
  __shared__ u16 lsK[64][256];
  __shared__ u16 lsVt[128][64];  // transposed: [e][k]
  __shared__ u16 lsP[4][16][64];
  const int tid = threadIdx.x, lane = tid & 31, w = tid >> 5;
  const int qblk = blockIdx.x, h = blockIdx.y, b = blockIdx.z;
  const size_t bh = (size_t)(b * Hsz + h);
  {
    const u16* gq =
        Qc + ((bh * Ssz) + qblk * 64 + (tid >> 1)) * 256 + (tid & 1) * 128;
    u16* dq = &lsQ[tid >> 1][(tid & 1) * 128];
#pragma unroll
    for (int c = 0; c < 16; ++c) async_ld_b128(&gq[c * 8], &dq[c * 8]);
  }
  wait_async();
  __syncthreads();
  float mrow[8], lrow[8];
#pragma unroll
  for (int r = 0; r < 8; ++r) { mrow[r] = -1e30f; lrow[r] = 0.f; }
  f32x8 oacc[8] = {};
  const int cLane = lane & 15;
  const int rHi = (lane >> 4) * 8;
  const float rs = 0.08838834764831845f;  // 1/sqrt(E)

  for (int k0 = 0; k0 < Ssz; k0 += 64) {
    const u16* gkTile = Kc + ((bh * Ssz) + k0) * 256;
    if (HAS_TDM) {
      if (w == 0) tdm_load_contig(gkTile, &lsK[0][0], 64 * 256 * 2);
    } else {
      const u16* gk = gkTile + (size_t)(tid >> 1) * 256 + (tid & 1) * 128;
      u16* dk = &lsK[tid >> 1][(tid & 1) * 128];
#pragma unroll
      for (int c = 0; c < 16; ++c) async_ld_b128(&gk[c * 8], &dk[c * 8]);
    }
    {
      const u16* gv = Vh + ((bh * Ssz) + k0) * 128 + tid;
#pragma unroll 4
      for (int kk = 0; kk < 64; ++kk) lsVt[tid][kk] = gv[(size_t)kk * 128];
    }
    if (HAS_TDM && w == 0) wait_tensor();
    wait_async();
    __syncthreads();
    // scores: 16 q-rows x 64 k-cols, inner dim 256
    f32x8 sacc[4] = {};
#pragma unroll
    for (int kk = 0; kk < 8; ++kk) {
      bf16x16 a = fragA(&lsQ[w * 16][0], 256, kk * 32, lane);
#pragma unroll
      for (int j = 0; j < 4; ++j) {
        bf16x16 bb = fragB(&lsK[j * 16][0], 256, kk * 32, lane);
        sacc[j] = wmma_bf16(a, bb, sacc[j]);
      }
    }
#pragma unroll
    for (int j = 0; j < 4; ++j) {
      const int kcol = k0 + j * 16 + cLane;
      const bool mok = mask[(size_t)b * Ssz + kcol] != 0;
#pragma unroll
      for (int r = 0; r < 8; ++r) {
        const float v = sacc[j][r] * rs;
        sacc[j][r] = mok ? v : -1e30f;
      }
    }
    // online softmax; row stats replicated across each 16-lane group
#pragma unroll
    for (int r = 0; r < 8; ++r) {
      float t = fmaxf(fmaxf(sacc[0][r], sacc[1][r]),
                      fmaxf(sacc[2][r], sacc[3][r]));
      t = fmaxf(t, __shfl_xor(t, 1, 32));
      t = fmaxf(t, __shfl_xor(t, 2, 32));
      t = fmaxf(t, __shfl_xor(t, 4, 32));
      t = fmaxf(t, __shfl_xor(t, 8, 32));
      const float mn = fmaxf(mrow[r], t);
      const float fac = __expf(mrow[r] - mn);
      float ssum = 0.f;
#pragma unroll
      for (int j = 0; j < 4; ++j) {
        const float p = __expf(sacc[j][r] - mn);
        sacc[j][r] = p;
        ssum += p;
      }
      ssum += __shfl_xor(ssum, 1, 32);
      ssum += __shfl_xor(ssum, 2, 32);
      ssum += __shfl_xor(ssum, 4, 32);
      ssum += __shfl_xor(ssum, 8, 32);
      lrow[r] = lrow[r] * fac + ssum;
      mrow[r] = mn;
#pragma unroll
      for (int jo = 0; jo < 8; ++jo) oacc[jo][r] *= fac;
    }
    // stash P as bf16 in per-wave LDS (LDS ops are in-order within a wave)
#pragma unroll
    for (int j = 0; j < 4; ++j)
#pragma unroll
      for (int r = 0; r < 8; ++r)
        lsP[w][r + rHi][j * 16 + cLane] = f2b(sacc[j][r]);
    // O += P * V
#pragma unroll
    for (int kk2 = 0; kk2 < 2; ++kk2) {
      bf16x16 a = fragA(&lsP[w][0][0], 64, kk2 * 32, lane);
#pragma unroll
      for (int jo = 0; jo < 8; ++jo) {
        bf16x16 bb = fragB(&lsVt[jo * 16][0], 64, kk2 * 32, lane);
        oacc[jo] = wmma_bf16(a, bb, oacc[jo]);
      }
    }
    __syncthreads();
  }
#pragma unroll
  for (int r = 0; r < 8; ++r) {
    const float rl = 1.f / (lrow[r] + 1e-20f);
    const int qrow = qblk * 64 + w * 16 + r + rHi;
    u16* orow = O + ((size_t)b * Ssz + qrow) * 1024 + h * 128;
#pragma unroll
    for (int jo = 0; jo < 8; ++jo)
      orow[jo * 16 + cLane] = f2b(oacc[jo][r] * rl);
  }
}

// ---------------------------------------------------------------------------
// LayerNorm over rows of 1024; writes bf16 (outb) or fp32 (outf).
// ---------------------------------------------------------------------------
__global__ __launch_bounds__(256) void ln_kernel(
    const float* __restrict__ x, const float* __restrict__ g,
    const float* __restrict__ bb, u16* __restrict__ outb,
    float* __restrict__ outf) {
  __shared__ float red[8];
  const int tid = threadIdx.x, lane = tid & 31, wid = tid >> 5;
  const size_t row = blockIdx.x;
  const float* xr = x + row * 1024;
  float v[4];
  float s = 0.f;
#pragma unroll
  for (int i = 0; i < 4; ++i) { v[i] = xr[tid + i * 256]; s += v[i]; }
#pragma unroll
  for (int m = 16; m >= 1; m >>= 1) s += __shfl_xor(s, m, 32);
  if (lane == 0) red[wid] = s;
  __syncthreads();
  float mean = 0.f;
#pragma unroll
  for (int i = 0; i < 8; ++i) mean += red[i];
  mean *= (1.f / 1024.f);
  __syncthreads();
  float vs = 0.f;
#pragma unroll
  for (int i = 0; i < 4; ++i) { const float d = v[i] - mean; vs += d * d; }
#pragma unroll
  for (int m = 16; m >= 1; m >>= 1) vs += __shfl_xor(vs, m, 32);
  if (lane == 0) red[wid] = vs;
  __syncthreads();
  float var = 0.f;
#pragma unroll
  for (int i = 0; i < 8; ++i) var += red[i];
  const float inv = rsqrtf(var * (1.f / 1024.f) + 1e-5f);
#pragma unroll
  for (int i = 0; i < 4; ++i) {
    const int c = tid + i * 256;
    const float o = (v[i] - mean) * inv * g[c] + bb[c];
    if (outb) outb[row * 1024 + c] = f2b(o);
    else outf[row * 1024 + c] = o;
  }
}

// ---------------------------------------------------------------------------
// Pack q,k,v (fp32, head-major cols) into attention operands:
//   Qc[.., 0:128]   = q + bu            (content query)
//   Qc[.., 128:256] = rot_q(q + bv)     (positional query, q-angle rotation)
//   Kc[.., 0:128]   = k
//   Kc[.., 128:256] = [sin(k/w_f), cos(k/w_f)]  (positional key table)
// ---------------------------------------------------------------------------
__global__ void qkv_pack_kernel(const float* __restrict__ qf,
                                const float* __restrict__ kf,
                                const float* __restrict__ vf,
                                const float* __restrict__ buL,
                                const float* __restrict__ bvL,
                                u16* __restrict__ Qc, u16* __restrict__ Kc,
                                u16* __restrict__ Vh) {
  const int idx = blockIdx.x * blockDim.x + threadIdx.x;
  if (idx >= Bsz * Ssz * Hsz * 64) return;
  const int f = idx & 63;
  const int h = (idx >> 6) & 7;
  const int t = idx >> 9;
  const int s = t & (Ssz - 1);
  const int b = t >> 10;
  const int e0 = 2 * f, e1 = 2 * f + 1;
  const size_t rowOff = (size_t)t * 1024 + h * 128;
  const float q0 = qf[rowOff + e0], q1 = qf[rowOff + e1];
  const float qu0 = q0 + buL[e0 * 8 + h], qu1 = q1 + buL[e1 * 8 + h];
  const float qv0 = q0 + bvL[e0 * 8 + h], qv1 = q1 + bvL[e1 * 8 + h];
  const float inv = __powf(10000.f, (float)e0 * (1.f / 128.f));
  const float ang = (float)s / inv;
  float sq, cq;
  __sincosf(ang, &sq, &cq);
  const size_t qr = ((size_t)(b * Hsz + h) * Ssz + s) * 256;
  Qc[qr + e0] = f2b(qu0);
  Qc[qr + e1] = f2b(qu1);
  Qc[qr + 128 + e0] = f2b(qv0 * cq + qv1 * sq);
  Qc[qr + 128 + e1] = f2b(qv1 * cq - qv0 * sq);
  Kc[qr + e0] = f2b(kf[rowOff + e0]);
  Kc[qr + e1] = f2b(kf[rowOff + e1]);
  Kc[qr + 128 + e0] = f2b(sq);
  Kc[qr + 128 + e1] = f2b(cq);
  const size_t vr = ((size_t)(b * Hsz + h) * Ssz + s) * 128;
  Vh[vr + e0] = f2b(vf[rowOff + e0]);
  Vh[vr + e1] = f2b(vf[rowOff + e1]);
}

// ---------------------------------------------------------------------------
// Weight conversions: all GEMM B operands stored TRANSPOSED, [N][K] bf16.
// ---------------------------------------------------------------------------
__global__ void conv_qkvT_kernel(const float* __restrict__ w,
                                 u16* __restrict__ o, int total) {
  // in: [L][M][E][H]  -> out: [L][N=h*128+e][K=m]
  for (int idx = blockIdx.x * blockDim.x + threadIdx.x; idx < total;
       idx += gridDim.x * blockDim.x) {
    const int h = idx & 7, e = (idx >> 3) & 127;
    const int m = (idx >> 10) & 1023, l = idx >> 20;
    o[((size_t)l * 1024 + h * 128 + e) * 1024 + m] = f2b(w[idx]);
  }
}

__global__ void conv_woT_kernel(const float* __restrict__ w,
                                u16* __restrict__ o, int total) {
  // in: [L][E][H][M] -> out: [L][N=m][K=h*128+e]
  for (int idx = blockIdx.x * blockDim.x + threadIdx.x; idx < total;
       idx += gridDim.x * blockDim.x) {
    const int m = idx & 1023, h = (idx >> 10) & 7;
    const int e = (idx >> 13) & 127, l = idx >> 20;
    o[((size_t)l * 1024 + m) * 1024 + h * 128 + e] = f2b(w[idx]);
  }
}

__global__ void conv_w1T_kernel(const float* __restrict__ w,
                                u16* __restrict__ o, int total) {
  // in: [L][M][I] -> out: [L][N=i][K=m]
  for (int idx = blockIdx.x * blockDim.x + threadIdx.x; idx < total;
       idx += gridDim.x * blockDim.x) {
    const int i = idx & 2047, m = (idx >> 11) & 1023, l = idx >> 21;
    o[((size_t)l * 2048 + i) * 1024 + m] = f2b(w[idx]);
  }
}

__global__ void conv_w2T_kernel(const float* __restrict__ w,
                                u16* __restrict__ o, int total) {
  // in: [L][I][M] -> out: [L][N=m][K=i]
  for (int idx = blockIdx.x * blockDim.x + threadIdx.x; idx < total;
       idx += gridDim.x * blockDim.x) {
    const int m = idx & 1023, i = (idx >> 10) & 2047, l = idx >> 21;
    o[((size_t)l * 1024 + m) * 2048 + i] = f2b(w[idx]);
  }
}

__global__ void copy_f32_kernel(const float* __restrict__ a,
                                float* __restrict__ o, int total) {
  for (int idx = blockIdx.x * blockDim.x + threadIdx.x; idx < total;
       idx += gridDim.x * blockDim.x)
    o[idx] = a[idx];
}

__global__ void add_inplace_kernel(float* __restrict__ x,
                                   const float* __restrict__ y,
                                   const float* __restrict__ bias, int colMask,
                                   int total) {
  for (int idx = blockIdx.x * blockDim.x + threadIdx.x; idx < total;
       idx += gridDim.x * blockDim.x)
    x[idx] += y[idx] + (bias ? bias[idx & colMask] : 0.f);
}

__global__ void relu_pack_kernel(const float* __restrict__ hf,
                                 const float* __restrict__ b1,
                                 u16* __restrict__ hb, int colMask, int total) {
  for (int idx = blockIdx.x * blockDim.x + threadIdx.x; idx < total;
       idx += gridDim.x * blockDim.x)
    hb[idx] = f2b(fmaxf(hf[idx] + b1[idx & colMask], 0.f));
}

// ---------------------------------------------------------------------------
extern "C" void kernel_launch(void* const* d_in, const int* in_sizes, int n_in,
                              void* d_out, int out_size, void* d_ws,
                              size_t ws_size, hipStream_t stream) {
  (void)in_sizes; (void)n_in; (void)out_size; (void)ws_size;
  const float* x = (const float*)d_in[0];
  const unsigned char* mask = (const unsigned char*)d_in[1];
  const float* wq = (const float*)d_in[2];
  const float* wk = (const float*)d_in[3];
  const float* wv = (const float*)d_in[4];
  const float* bu = (const float*)d_in[5];
  const float* bv = (const float*)d_in[6];
  const float* wo = (const float*)d_in[7];
  const float* an_g = (const float*)d_in[8];
  const float* an_b = (const float*)d_in[9];
  const float* w1 = (const float*)d_in[10];
  const float* b1 = (const float*)d_in[11];
  const float* w2 = (const float*)d_in[12];
  const float* b2 = (const float*)d_in[13];
  const float* fn_g = (const float*)d_in[14];
  const float* fn_b = (const float*)d_in[15];
  const float* out_g = (const float*)d_in[16];
  const float* out_b = (const float*)d_in[17];

  char* ws = (char*)d_ws;
  size_t off = 0;
  auto alloc = [&](size_t bytes) {
    char* p = ws + off;
    off += (bytes + 255) & ~(size_t)255;
    return p;
  };
  const size_t MB = 1024 * 1024;
  u16* wqb = (u16*)alloc((size_t)Lsz * Msz * Msz * 2);
  u16* wkb = (u16*)alloc((size_t)Lsz * Msz * Msz * 2);
  u16* wvb = (u16*)alloc((size_t)Lsz * Msz * Msz * 2);
  u16* wob = (u16*)alloc((size_t)Lsz * Msz * Msz * 2);
  u16* w1b = (u16*)alloc((size_t)Lsz * Msz * Isz * 2);
  u16* w2b = (u16*)alloc((size_t)Lsz * Msz * Isz * 2);
  float* xcur = (float*)alloc((size_t)Bsz * Ssz * Msz * 4);
  u16* ybf = (u16*)alloc((size_t)Bsz * Ssz * Msz * 2);
  char* scratch = alloc(96 * MB);  // qf|kf|vf overlap hf|gout across phases
  float* qf = (float*)scratch;
  float* kf = (float*)(scratch + 32 * MB);
  float* vf = (float*)(scratch + 64 * MB);
  float* hf = (float*)scratch;               // FFN hidden fp32 (64 MB)
  float* gout = (float*)(scratch + 64 * MB); // GEMM out (32 MB)
  u16* Qc = (u16*)alloc((size_t)Bsz * Hsz * Ssz * 256 * 2);
  u16* Kc = (u16*)alloc((size_t)Bsz * Hsz * Ssz * 256 * 2);
  u16* Vh = (u16*)alloc((size_t)Bsz * Hsz * Ssz * 128 * 2);
  u16* Ob = (u16*)alloc((size_t)Bsz * Ssz * Msz * 2);
  u16* hbf = (u16*)alloc((size_t)Bsz * Ssz * Isz * 2);

  const int T = 256;
  conv_qkvT_kernel<<<4096, T, 0, stream>>>(wq, wqb, Lsz * Msz * Msz);
  conv_qkvT_kernel<<<4096, T, 0, stream>>>(wk, wkb, Lsz * Msz * Msz);
  conv_qkvT_kernel<<<4096, T, 0, stream>>>(wv, wvb, Lsz * Msz * Msz);
  conv_woT_kernel<<<4096, T, 0, stream>>>(wo, wob, Lsz * Msz * Msz);
  conv_w1T_kernel<<<4096, T, 0, stream>>>(w1, w1b, Lsz * Msz * Isz);
  conv_w2T_kernel<<<4096, T, 0, stream>>>(w2, w2b, Lsz * Msz * Isz);
  copy_f32_kernel<<<4096, T, 0, stream>>>(x, xcur, Bsz * Ssz * Msz);

  const int rows = Bsz * Ssz;  // 8192
  const dim3 gN1024(Msz / 128, rows / 128);
  const dim3 gN2048(Isz / 128, rows / 128);

  for (int l = 0; l < Lsz; ++l) {
    ln_kernel<<<rows, T, 0, stream>>>(xcur, an_g + l * Msz, an_b + l * Msz,
                                      ybf, nullptr);
    gemm_bf16_kernel<<<gN1024, T, 0, stream>>>(
        ybf, wqb + (size_t)l * Msz * Msz, qf, nullptr, Msz, Msz);
    gemm_bf16_kernel<<<gN1024, T, 0, stream>>>(
        ybf, wkb + (size_t)l * Msz * Msz, kf, nullptr, Msz, Msz);
    gemm_bf16_kernel<<<gN1024, T, 0, stream>>>(
        ybf, wvb + (size_t)l * Msz * Msz, vf, nullptr, Msz, Msz);
    qkv_pack_kernel<<<(Bsz * Ssz * Hsz * 64) / T, T, 0, stream>>>(
        qf, kf, vf, bu + l * Esz * Hsz, bv + l * Esz * Hsz, Qc, Kc, Vh);
    attn_kernel<<<dim3(Ssz / 64, Hsz, Bsz), 128, 0, stream>>>(Qc, Kc, Vh, mask,
                                                              Ob);
    gemm_bf16_kernel<<<gN1024, T, 0, stream>>>(
        Ob, wob + (size_t)l * Msz * Msz, gout, nullptr, Msz, Msz);
    add_inplace_kernel<<<4096, T, 0, stream>>>(xcur, gout, nullptr, Msz - 1,
                                               rows * Msz);
    ln_kernel<<<rows, T, 0, stream>>>(xcur, fn_g + l * Msz, fn_b + l * Msz,
                                      ybf, nullptr);
    gemm_bf16_kernel<<<gN2048, T, 0, stream>>>(
        ybf, w1b + (size_t)l * Msz * Isz, hf, nullptr, Isz, Msz);
    relu_pack_kernel<<<4096, T, 0, stream>>>(hf, b1 + l * Isz, hbf, Isz - 1,
                                             rows * Isz);
    gemm_bf16_kernel<<<gN1024, T, 0, stream>>>(
        hbf, w2b + (size_t)l * Isz * Msz, gout, nullptr, Msz, Isz);
    add_inplace_kernel<<<4096, T, 0, stream>>>(xcur, gout, b2 + l * Msz,
                                               Msz - 1, rows * Msz);
  }
  ln_kernel<<<rows, T, 0, stream>>>(xcur, out_g, out_b, nullptr,
                                    (float*)d_out);
}
